// Mlp_59983513256146
// MI455X (gfx1250) — 8-bit matrix operand order repaired and checked by reading, compiled, not run on hardware
//
#include <hip/hip_runtime.h>
#include <cstdint>
#include <cstddef>


typedef __attribute__((ext_vector_type(8))) int v8i;

#define QMAXF 127.0f


__global__ void hmq_init_scal(unsigned int* s) {
  if (threadIdx.x < 4) s[threadIdx.x] = 0u;
}

__global__ __launch_bounds__(256)
void hmq_absmax_f32(const float4* __restrict__ x, long long n4, unsigned int* out) {
  __shared__ float red[256];
  float m = 0.0f;
  long long i = (long long)blockIdx.x * blockDim.x + threadIdx.x;
  const long long stride = (long long)gridDim.x * blockDim.x;
  for (; i < n4; i += stride) {
    float4 v = x[i];
    m = fmaxf(m, fmaxf(fmaxf(fabsf(v.x), fabsf(v.y)),
                       fmaxf(fabsf(v.z), fabsf(v.w))));
  }
  red[threadIdx.x] = m;
  __syncthreads();
  for (int o = 128; o > 0; o >>= 1) {
    if ((int)threadIdx.x < o)
      red[threadIdx.x] = fmaxf(red[threadIdx.x], red[threadIdx.x + o]);
    __syncthreads();
  }
  if (threadIdx.x == 0) atomicMax(out, __float_as_uint(red[0]));
}

__device__ __forceinline__ signed char quant1(float v, float inv) {
  return (signed char)(int)fminf(fmaxf(rintf(v * inv), -QMAXF), QMAXF);
}

__global__ __launch_bounds__(256)
void hmq_quant_i8(const float4* __restrict__ x, long long n4,
                  const unsigned int* __restrict__ amax, char4* __restrict__ q) {
  const float s = fmaxf(__uint_as_float(*amax), 1e-8f) * (1.0f / QMAXF);
  const float inv = 1.0f / s;
  long long i = (long long)blockIdx.x * blockDim.x + threadIdx.x;
  const long long stride = (long long)gridDim.x * blockDim.x;
  for (; i < n4; i += stride) {
    float4 v = x[i];
    char4 o;
    o.x = quant1(v.x, inv);
    o.y = quant1(v.y, inv);
    o.z = quant1(v.z, inv);
    o.w = quant1(v.w, inv);
    q[i] = o;
  }
}


__device__ __forceinline__ void async_g2l_b128(const signed char* lds_dst,
                                               const signed char* g_src) {
  unsigned lds_off = (unsigned)(uintptr_t)lds_dst;
  unsigned long long ga = (unsigned long long)(uintptr_t)g_src;
  asm volatile("global_load_async_to_lds_b128 %0, %1, off"
               :: "v"(lds_off), "v"(ga)
               : "memory");
}

__device__ __forceinline__ void wait_asynccnt0() {
#if __has_builtin(__builtin_amdgcn_s_wait_asynccnt)
  __builtin_amdgcn_s_wait_asynccnt(0);
#else
  asm volatile("s_wait_asynccnt 0x0" ::: "memory");
#endif
}


#define MBLK 128
#define NBLK 128
#define KSTEP 128
#define ASTR 144
#define BSTR 144

template<bool GELU, bool AMAX>
__global__ __launch_bounds__(256)
void hmq_gemm_iu8(const signed char* __restrict__ A,
                  const signed char* __restrict__ B,
                  const float* __restrict__ bias,
                  const unsigned int* scal, int sa_idx, int sb_idx,
                  float* __restrict__ C,
                  unsigned int* amax_out,
                  int M, int N, int K) {
  __shared__ __align__(16) signed char lA[2][MBLK * ASTR];
  __shared__ __align__(16) signed char lB[2][NBLK * BSTR];

  const int tid  = threadIdx.x;
  const int lane = tid & 31;
  const int wave = tid >> 5;
  const int wm   = wave >> 2;
  const int wn   = wave & 3;
  const int l15  = lane & 15;
  const int hi   = lane >> 4;

  const int m0 = blockIdx.x * MBLK;
  const int n0 = blockIdx.y * NBLK;

  const float sa  = fmaxf(__uint_as_float(scal[sa_idx]), 1e-8f) * (1.0f / QMAXF);
  const float sb  = fmaxf(__uint_as_float(scal[sb_idx]), 1e-8f) * (1.0f / QMAXF);
  const float sab = sa * sb;

  v8i acc[4][2];
#pragma unroll
  for (int i = 0; i < 4; ++i)
#pragma unroll
    for (int j = 0; j < 2; ++j)
#pragma unroll
      for (int r = 0; r < 8; ++r) acc[i][j][r] = 0;

  const int lr  = tid >> 1;
  const int lc0 = (tid & 1) * 64;

  const signed char* gA = A + (size_t)(m0 + lr) * K + lc0;
  const signed char* gB = B + (size_t)(n0 + lr) * K + lc0;

  auto issue_tile = [&](int buf, int k0) {
#pragma unroll
    for (int j = 0; j < 4; ++j)
      async_g2l_b128(&lA[buf][lr * ASTR + lc0 + j * 16], gA + k0 + j * 16);
#pragma unroll
    for (int j = 0; j < 4; ++j)
      async_g2l_b128(&lB[buf][lr * BSTR + lc0 + j * 16], gB + k0 + j * 16);
  };

  issue_tile(0, 0);
  wait_asynccnt0();
  __syncthreads();

  int cur = 0;
  for (int k0 = 0; k0 < K; k0 += KSTEP) {
    const int nxt = cur ^ 1;
    if (k0 + KSTEP < K) issue_tile(nxt, k0 + KSTEP);

    v8i bfrag[2][2];
#pragma unroll
    for (int sn = 0; sn < 2; ++sn)
#pragma unroll
      for (int kh = 0; kh < 2; ++kh) {
        const signed char* p =
            &lB[cur][(wn * 32 + sn * 16 + l15) * BSTR + kh * 64 + hi * 16];
        int4 c0 = *(const int4*)(p);
        int4 c1 = *(const int4*)(p + 32);
        v8i b;
        b[0] = c0.x; b[1] = c0.y; b[2] = c0.z; b[3] = c0.w;
        b[4] = c1.x; b[5] = c1.y; b[6] = c1.z; b[7] = c1.w;
        bfrag[sn][kh] = b;
      }

    v8i afrag[4][2];
#pragma unroll
    for (int sm = 0; sm < 4; ++sm)
#pragma unroll
      for (int kh = 0; kh < 2; ++kh) {
        const signed char* p =
            &lA[cur][(wm * 64 + sm * 16 + l15) * ASTR + kh * 64 + hi * 16];
        int2 c0 = *(const int2*)(p);
        int2 c1 = *(const int2*)(p + 8);
        int2 c2 = *(const int2*)(p + 32);
        int2 c3 = *(const int2*)(p + 40);
        v8i a;
        a[0] = c0.x; a[1] = c0.y; a[2] = c1.x; a[3] = c1.y;
        a[4] = c2.x; a[5] = c2.y; a[6] = c3.x; a[7] = c3.y;
        afrag[sm][kh] = a;
      }

#pragma unroll
    for (int sm = 0; sm < 4; ++sm)
#pragma unroll
      for (int sn = 0; sn < 2; ++sn) {
        acc[sm][sn] = __builtin_amdgcn_wmma_i32_16x16x64_iu8(
            true, afrag[sm][0], true, bfrag[sn][0], acc[sm][sn], false, false);
        acc[sm][sn] = __builtin_amdgcn_wmma_i32_16x16x64_iu8(
            true, afrag[sm][1], true, bfrag[sn][1], acc[sm][sn], false, false);
      }

    wait_asynccnt0();
    __syncthreads();
    cur = nxt;
  }

  float lmax = 0.0f;
#pragma unroll
  for (int sm = 0; sm < 4; ++sm) {
    const int rowb = m0 + wm * 64 + sm * 16 + hi * 8;
#pragma unroll
    for (int sn = 0; sn < 2; ++sn) {
      const int col = n0 + wn * 32 + sn * 16 + l15;
      const float bv = bias[col];
#pragma unroll
      for (int r = 0; r < 8; ++r) {
        float v = (float)acc[sm][sn][r] * sab + bv;
        if (GELU) v = 0.5f * v * (1.0f + erff(v * 0.7071067811865476f));
        C[(size_t)(rowb + r) * N + col] = v;
        if (AMAX) lmax = fmaxf(lmax, fabsf(v));
      }
    }
  }
  if (AMAX) atomicMax(amax_out, __float_as_uint(lmax));
}


static inline uintptr_t align256(uintptr_t v) {
  return (v + 255) & ~(uintptr_t)255;
}

extern "C" void kernel_launch(void* const* d_in, const int* in_sizes, int n_in,
                              void* d_out, int out_size, void* d_ws, size_t ws_size,
                              hipStream_t stream) {
  const float* x  = (const float*)d_in[0];
  const float* w1 = (const float*)d_in[1];
  const float* b1 = (const float*)d_in[2];
  const float* w2 = (const float*)d_in[3];
  const float* b2 = (const float*)d_in[4];

  const long long H = in_sizes[2];
  const long long D = in_sizes[4];
  const long long M = (long long)in_sizes[0] / D;

  uintptr_t cur = (uintptr_t)d_ws;
  unsigned int* scal = (unsigned int*)cur;
  cur = align256(cur + 4 * sizeof(unsigned int));
  signed char* xq  = (signed char*)cur; cur = align256(cur + (size_t)(M * D));
  signed char* w1q = (signed char*)cur; cur = align256(cur + (size_t)(H * D));
  signed char* w2q = (signed char*)cur; cur = align256(cur + (size_t)(D * H));
  signed char* gq  = (signed char*)cur; cur = align256(cur + (size_t)(M * H));
  float*       g   = (float*)cur;

  const int RB = 2048;

  hmq_init_scal<<<1, 64, 0, stream>>>(scal);

  hmq_absmax_f32<<<RB, 256, 0, stream>>>((const float4*)x,  M * D / 4, scal + 0);
  hmq_absmax_f32<<<RB, 256, 0, stream>>>((const float4*)w1, H * D / 4, scal + 1);
  hmq_absmax_f32<<<RB, 256, 0, stream>>>((const float4*)w2, D * H / 4, scal + 2);

  hmq_quant_i8<<<RB, 256, 0, stream>>>((const float4*)x,  M * D / 4, scal + 0, (char4*)xq);
  hmq_quant_i8<<<RB, 256, 0, stream>>>((const float4*)w1, H * D / 4, scal + 1, (char4*)w1q);
  hmq_quant_i8<<<RB, 256, 0, stream>>>((const float4*)w2, D * H / 4, scal + 2, (char4*)w2q);

  {
    dim3 grid((unsigned)(M / MBLK), (unsigned)(H / NBLK));
    hmq_gemm_iu8<true, true><<<grid, 256, 0, stream>>>(
        xq, w1q, b1, scal, 0, 1, g, scal + 3, (int)M, (int)H, (int)D);
  }

  hmq_quant_i8<<<RB, 256, 0, stream>>>((const float4*)g, M * H / 4, scal + 3, (char4*)gq);

  {
    dim3 grid((unsigned)(M / MBLK), (unsigned)(D / NBLK));
    hmq_gemm_iu8<false, false><<<grid, 256, 0, stream>>>(
        gq, w2q, b2, scal, 3, 2, (float*)d_out, nullptr, (int)M, (int)D, (int)H);
  }
}
